// EasySlidingWindowAttention_75831942578242
// MI455X (gfx1250) — matrix operand order repaired and checked by reading, not compiled, not run on hardware
//
#include <hip/hip_runtime.h>
#include <hip/hip_bf16.h>
#include <math.h>

// ---------- problem constants ----------
#define B_   2
#define C_   2048
#define E_   1024
#define H_   16
#define HD_  64
#define W_   128
#define M_   (B_*C_)          // 4096 rows

// GEMM blocking
#define BM 128
#define BN 128
#define BK 32
#define TILEE (BM*BK)         // halves per LDS tile (4096 bf16 = 8KB)

// ---------- types ----------
typedef __bf16 bf16;
typedef __bf16 v16bf __attribute__((ext_vector_type(16)));
typedef __bf16 v8bf  __attribute__((ext_vector_type(8)));
typedef float  v8f   __attribute__((ext_vector_type(8)));
typedef unsigned int       u32;
typedef unsigned long long u64;
typedef u32 u32x4 __attribute__((ext_vector_type(4)));
typedef int i32x4 __attribute__((ext_vector_type(4)));
typedef int i32x8 __attribute__((ext_vector_type(8)));

static __device__ __forceinline__ v16bf cat16(v8bf lo, v8bf hi) {
  return __builtin_shufflevector(lo, hi, 0,1,2,3,4,5,6,7,8,9,10,11,12,13,14,15);
}

// A-matrix 16x32 bf16 fragment: lanes 0-15 hold M=lane, K runs
// {0..15}; lanes 16-31 hold M=lane-16, K runs {16..31}.
static __device__ __forceinline__ v16bf load_a_frag(const bf16* rowp, int klo) {
  v8bf lo = *(const v8bf*)(rowp + 2 * klo);
  v8bf hi = *(const v8bf*)(rowp + 2 * klo + 8);
  return cat16(lo, hi);
}

static __device__ __forceinline__ v8f wmma_bf16(v16bf a, v16bf b, v8f c) {
  return __builtin_amdgcn_wmma_f32_16x16x32_bf16(false, a, false, b, (short)0, c, false, false);
}

// ---------- Tensor Data Mover: 2-D tile global -> LDS ----------
// D# per CDNA5 ISA: group0 = {count/type/lds/global addr},
// group1 = {data_size, tensor dims, tile dims, stride}. Groups 2/3 zero (2-D).
static __device__ __forceinline__ void tdm_load_2d(
    unsigned lds_byte_addr, const void* gptr,
    u32 tile_d0 /*elems*/, u32 tile_d1 /*rows*/, u64 row_stride /*elems*/) {
  u64 ga = (u64)(size_t)gptr;
  u32x4 g0;
  g0.x = 1u;                                   // count=1, user mode, no gather
  g0.y = lds_byte_addr;                        // bits 63:32  lds_addr
  g0.z = (u32)(ga & 0xffffffffu);              // bits 95:64  global_addr lo
  g0.w = (u32)((ga >> 32) & 0x01ffffffu) | (2u << 30); // addr[56:32] | type=2
  const u32 td0 = 1u << 20, td1 = 1u << 20;    // huge tensor dims: no OOB clip
  i32x8 g1;
  g1[0] = (int)(1u << 16);                     // data_size=1 (2 bytes)
  g1[1] = (int)((td0 & 0xffffu) << 16);        // tensor_dim0[15:0] at bit 48
  g1[2] = (int)((td0 >> 16) | ((td1 & 0xffffu) << 16));
  g1[3] = (int)((td1 >> 16) | ((tile_d0 & 0xffffu) << 16)); // tile_dim0
  g1[4] = (int)(tile_d1 & 0xffffu);            // tile_dim1 (tile_dim2 = 0)
  g1[5] = (int)(u32)(row_stride & 0xffffffffu);          // dim0_stride lo
  g1[6] = (int)(u32)((row_stride >> 32) & 0xffffu);      // dim0_stride hi
  g1[7] = 0;
  i32x4 z4 = {0, 0, 0, 0};
#if defined(__clang_major__) && (__clang_major__ >= 23)
  i32x8 z8 = {0, 0, 0, 0, 0, 0, 0, 0};
  __builtin_amdgcn_tensor_load_to_lds(g0, g1, z4, z4, z8, 0);
#else
  __builtin_amdgcn_tensor_load_to_lds(g0, g1, z4, z4, 0);
#endif
}

static __device__ __forceinline__ unsigned lds_off(const void* p) {
  return (unsigned)(size_t)p;   // low 32 bits of flat LDS address = byte offset
}

// ---------- fp32 -> bf16 convert ----------
__global__ void cvt_f32_bf16(const float* __restrict__ in, bf16* __restrict__ out, int n) {
  int i = blockIdx.x * blockDim.x + threadIdx.x;
  if (i < n) out[i] = (bf16)in[i];
}

// ---------- TDM-fed double-buffered block GEMM core ----------
// Block: 8 waves, tile BM(128) x BN(128); wave (wm 0..3, wn 0..1) owns 32x64.
// acc[mi][si]: mi = 16-row subtile, si = 16-col subtile.
static __device__ __forceinline__ void gemm_core(
    const bf16* __restrict__ A, const bf16* __restrict__ Wt, int K,
    int m0, int n0, bf16* sA, bf16* sB, v8f acc[2][4]) {
  int lane = threadIdx.x & 31, wave = threadIdx.x >> 5;
  int wm = wave >> 1, wn = wave & 1;
  int lm  = lane & 15;
  int klo = (lane & 16) ? 8  : 0;
  int khi = (lane & 16) ? 16 : 0;

  if (wave == 0) {   // prologue: stage buffer 0 via TDM
    tdm_load_2d(lds_off(sA), A  + (size_t)m0 * K, BK, BM, (u64)K);
    tdm_load_2d(lds_off(sB), Wt + (size_t)n0 * K, BK, BN, (u64)K);
    __builtin_amdgcn_s_wait_tensorcnt(0);
  }
  __syncthreads();

  int buf = 0;
  for (int kc = 0; kc < K; kc += BK) {
    int nbuf = buf ^ 1;
    if (wave == 0 && kc + BK < K) {   // prefetch next K-slice into other buffer
      tdm_load_2d(lds_off(sA + nbuf * TILEE), A  + (size_t)m0 * K + kc + BK, BK, BM, (u64)K);
      tdm_load_2d(lds_off(sB + nbuf * TILEE), Wt + (size_t)n0 * K + kc + BK, BK, BN, (u64)K);
    }
    const bf16* Ab = sA + buf * TILEE;
    const bf16* Bb = sB + buf * TILEE;
    v16bf af[2];
#pragma unroll
    for (int mi = 0; mi < 2; ++mi)
      af[mi] = load_a_frag(Ab + (wm * 32 + mi * 16 + lm) * BK, klo);
#pragma unroll
    for (int si = 0; si < 4; ++si) {
      v16bf bfg = *(const v16bf*)(Bb + (wn * 64 + si * 16 + lm) * BK + khi);
#pragma unroll
      for (int mi = 0; mi < 2; ++mi)
        acc[mi][si] = wmma_bf16(af[mi], bfg, acc[mi][si]);
    }
    if (wave == 0 && kc + BK < K) __builtin_amdgcn_s_wait_tensorcnt(0);
    __syncthreads();
    buf = nbuf;
  }
}

// ---------- QKV projection: (4096x1024) x (1024x3072) + bias ----------
__global__ void __launch_bounds__(256) gemm_qkv(
    const bf16* __restrict__ A, const bf16* __restrict__ Wt,
    const float* __restrict__ bias,
    bf16* __restrict__ Q, bf16* __restrict__ Kt, bf16* __restrict__ Vt) {
  __shared__ __align__(32) bf16 sA[2][TILEE];
  __shared__ __align__(32) bf16 sB[2][TILEE];
  const int K = E_, N = 3 * E_;
  const int nblk_n = N / BN;                  // 24
  int m0 = ((int)blockIdx.x / nblk_n) * BM;
  int n0 = ((int)blockIdx.x % nblk_n) * BN;
  v8f acc[2][4] = {};
  gemm_core(A, Wt, K, m0, n0, &sA[0][0], &sB[0][0], acc);

  int lane = threadIdx.x & 31, wave = threadIdx.x >> 5;
  int wm = wave >> 1, wn = wave & 1;
  int lm = lane & 15, half8 = (lane & 16) ? 8 : 0;
#pragma unroll
  for (int mi = 0; mi < 2; ++mi) {
#pragma unroll
    for (int si = 0; si < 4; ++si) {
      int n = n0 + wn * 64 + si * 16 + lm;
      float bv = bias[n];
      int which = n >> 10;                    // 0=q 1=k 2=v
      int nh = n & (E_ - 1);
      int h = nh >> 6, hd = nh & 63;
#pragma unroll
      for (int r = 0; r < 8; ++r) {
        int m = m0 + wm * 32 + mi * 16 + r + half8;
        int b = m >> 11, c = m & (C_ - 1);
        bf16 v = (bf16)(acc[mi][si][r] + bv);
        size_t bh = (size_t)(b * H_ + h);
        if (which == 0)      Q [(bh * C_ + c) * HD_ + hd] = v;
        else if (which == 1) Kt[(bh * C_ + c) * HD_ + hd] = v;
        else                 Vt[(bh * HD_ + hd) * C_ + c] = v;  // V transposed
      }
    }
  }
}

// ---------- output projection: (4096x1024) x (1024x1024) + bias -> fp32 ----------
__global__ void __launch_bounds__(256) gemm_out(
    const bf16* __restrict__ A, const bf16* __restrict__ Wt,
    const float* __restrict__ bias, float* __restrict__ out) {
  __shared__ __align__(32) bf16 sA[2][TILEE];
  __shared__ __align__(32) bf16 sB[2][TILEE];
  const int K = E_, N = E_;
  const int nblk_n = N / BN;                  // 8
  int m0 = ((int)blockIdx.x / nblk_n) * BM;
  int n0 = ((int)blockIdx.x % nblk_n) * BN;
  v8f acc[2][4] = {};
  gemm_core(A, Wt, K, m0, n0, &sA[0][0], &sB[0][0], acc);

  int lane = threadIdx.x & 31, wave = threadIdx.x >> 5;
  int wm = wave >> 1, wn = wave & 1;
  int lm = lane & 15, half8 = (lane & 16) ? 8 : 0;
#pragma unroll
  for (int mi = 0; mi < 2; ++mi) {
#pragma unroll
    for (int si = 0; si < 4; ++si) {
      int n = n0 + wn * 64 + si * 16 + lm;
      float bv = bias[n];
#pragma unroll
      for (int r = 0; r < 8; ++r) {
        int m = m0 + wm * 32 + mi * 16 + r + half8;
        out[(size_t)m * N + n] = acc[mi][si][r] + bv;
      }
    }
  }
}

// ---------- sliding-window flash attention ----------
// one wave per (b, h, 16-row q tile); 9 aligned 32-key chunks cover [i0-128, i0+144)
__global__ void __launch_bounds__(128) attn_swa(
    const bf16* __restrict__ Q, const bf16* __restrict__ Kh,
    const bf16* __restrict__ Vt, const unsigned char* __restrict__ am,
    bf16* __restrict__ O) {
  __shared__ __align__(16) bf16 pbuf[4][16][32];
  int lane = threadIdx.x & 31, wave = threadIdx.x >> 5;
  int tile = blockIdx.x * 4 + wave;        // 0..4095
  int qt = tile & 127;
  int bh = tile >> 7;
  int h = bh & (H_ - 1), b = bh >> 4;
  int i0 = qt * 16;
  int lm   = lane & 15;
  int klo  = (lane & 16) ? 8  : 0;
  int khi  = (lane & 16) ? 16 : 0;
  int half8 = (lane & 16) ? 8 : 0;
  const bf16* Qp = Q  + (size_t)bh * C_ * HD_;
  const bf16* Kp = Kh + (size_t)bh * C_ * HD_;
  const bf16* Vp = Vt + (size_t)bh * HD_ * C_;
  const unsigned char* amp = am + (size_t)b * C_ * C_;

  v16bf qf0 = load_a_frag(Qp + (size_t)(i0 + lm) * HD_, klo);
  v16bf qf1 = load_a_frag(Qp + (size_t)(i0 + lm) * HD_ + 32, klo);

  float mr[8], lr[8];
#pragma unroll
  for (int r = 0; r < 8; ++r) { mr[r] = -INFINITY; lr[r] = 0.f; }
  v8f o[4] = {};

  for (int ck = 0; ck < 9; ++ck) {
    int j0 = i0 - 128 + ck * 32;           // 16-aligned; sub-windows fully in/out
    float sv[2][8];
    float p[2][8];
    float cm[8];
#pragma unroll
    for (int r = 0; r < 8; ++r) cm[r] = -INFINITY;

#pragma unroll
    for (int sub = 0; sub < 2; ++sub) {
      int jb = j0 + sub * 16;
      int jbs = (jb >= 0 && jb + 16 <= C_) ? jb : 0;  // clamp; masked below
      const bf16* krow = Kp + (size_t)(jbs + lm) * HD_ + khi;
      v16bf bk0 = *(const v16bf*)(krow);
      v16bf bk1 = *(const v16bf*)(krow + 32);
      v8f s = {};
      s = wmma_bf16(qf0, bk0, s);
      s = wmma_bf16(qf1, bk1, s);
#pragma unroll
      for (int r = 0; r < 8; ++r) {
        int i = i0 + r + half8;
        int j = jb + lm;
        bool jok = (unsigned)j < (unsigned)C_;
        int d = i - j; d = d < 0 ? -d : d;
        bool band = (d >= 1 && d <= W_) || (d == 0 && (i == 0 || i == C_ - 1));
        bool ok = jok && band;
        if (ok) ok = amp[(size_t)i * C_ + j] != 0;
        float x = ok ? s[r] * 0.125f : -INFINITY;   // scale = 1/sqrt(64)
        sv[sub][r] = x;
        cm[r] = fmaxf(cm[r], x);
      }
    }
#pragma unroll
    for (int r = 0; r < 8; ++r) {
#pragma unroll
      for (int off = 1; off < 16; off <<= 1)
        cm[r] = fmaxf(cm[r], __shfl_xor(cm[r], off, 32));
    }
    float alpha[8], rs[8];
#pragma unroll
    for (int r = 0; r < 8; ++r) {
      float mn = fmaxf(mr[r], cm[r]);
      alpha[r] = (mn == mr[r]) ? 1.0f : __expf(mr[r] - mn);
      mr[r] = mn;
      float p0 = (sv[0][r] == -INFINITY) ? 0.f : __expf(sv[0][r] - mn);
      float p1 = (sv[1][r] == -INFINITY) ? 0.f : __expf(sv[1][r] - mn);
      p[0][r] = p0; p[1][r] = p1;
      rs[r] = p0 + p1;
    }
#pragma unroll
    for (int r = 0; r < 8; ++r) {
#pragma unroll
      for (int off = 1; off < 16; off <<= 1)
        rs[r] += __shfl_xor(rs[r], off, 32);
      lr[r] = lr[r] * alpha[r] + rs[r];
    }
#pragma unroll
    for (int s4 = 0; s4 < 4; ++s4)
#pragma unroll
      for (int r = 0; r < 8; ++r)
        o[s4][r] *= alpha[r];

    // transpose P (C/D layout) -> A layout through LDS
#pragma unroll
    for (int sub = 0; sub < 2; ++sub)
#pragma unroll
      for (int r = 0; r < 8; ++r)
        pbuf[wave][r + half8][sub * 16 + lm] = (bf16)p[sub][r];
    __syncthreads();
    v8bf plo = *(const v8bf*)&pbuf[wave][lm][2 * klo];
    v8bf phi = *(const v8bf*)&pbuf[wave][lm][2 * klo + 8];
    v16bf pa = cat16(plo, phi);

    int vs = j0 + khi;
    int vsafe = (vs >= 0 && vs + 16 <= C_) ? vs : 0;
#pragma unroll
    for (int sd = 0; sd < 4; ++sd) {
      const bf16* vp = Vp + (size_t)(sd * 16 + lm) * C_ + vsafe;
      v16bf vf = *(const v16bf*)vp;
      o[sd] = wmma_bf16(pa, vf, o[sd]);
    }
    __syncthreads();
  }

#pragma unroll
  for (int r = 0; r < 8; ++r) lr[r] = 1.0f / lr[r];
#pragma unroll
  for (int sd = 0; sd < 4; ++sd) {
#pragma unroll
    for (int r = 0; r < 8; ++r) {
      int i = i0 + r + half8;
      int col = h * HD_ + sd * 16 + lm;
      O[((size_t)b * C_ + i) * E_ + col] = (bf16)(o[sd][r] * lr[r]);
    }
  }
}

// ---------- launcher ----------
extern "C" void kernel_launch(void* const* d_in, const int* in_sizes, int n_in,
                              void* d_out, int out_size, void* d_ws, size_t ws_size,
                              hipStream_t stream) {
  (void)in_sizes; (void)n_in; (void)out_size; (void)ws_size;
  const float*         x  = (const float*)d_in[0];
  const unsigned char* am = (const unsigned char*)d_in[1];
  const float*         w1 = (const float*)d_in[2];
  const float*         b1 = (const float*)d_in[3];
  const float*         wo = (const float*)d_in[4];
  const float*         bo = (const float*)d_in[5];
  float* out = (float*)d_out;

  char* ws = (char*)d_ws;
  bf16* Xbf  = (bf16*)ws; ws += (size_t)M_ * E_ * 2;
  bf16* W1bf = (bf16*)ws; ws += (size_t)3 * E_ * E_ * 2;
  bf16* Wobf = (bf16*)ws; ws += (size_t)E_ * E_ * 2;
  bf16* Qb   = (bf16*)ws; ws += (size_t)B_ * H_ * C_ * HD_ * 2;
  bf16* Kb   = (bf16*)ws; ws += (size_t)B_ * H_ * C_ * HD_ * 2;
  bf16* Vtb  = (bf16*)ws; ws += (size_t)B_ * H_ * HD_ * C_ * 2;
  bf16* Ob   = (bf16*)ws; ws += (size_t)M_ * E_ * 2;

  cvt_f32_bf16<<<(M_ * E_) / 256, 256, 0, stream>>>(x,  Xbf,  M_ * E_);
  cvt_f32_bf16<<<(3 * E_ * E_) / 256, 256, 0, stream>>>(w1, W1bf, 3 * E_ * E_);
  cvt_f32_bf16<<<(E_ * E_) / 256, 256, 0, stream>>>(wo, Wobf, E_ * E_);

  // QKV: (4096/128)*(3072/128) = 32*24 = 768 blocks
  gemm_qkv<<<768, 256, 0, stream>>>(Xbf, W1bf, b1, Qb, Kb, Vtb);
  // attention: 2*16*128 = 4096 wave tiles, 4 waves/block
  attn_swa<<<1024, 128, 0, stream>>>(Qb, Kb, Vtb, am, Ob);
  // out proj: (4096/128)*(1024/128) = 32*8 = 256 blocks
  gemm_out<<<256, 256, 0, stream>>>(Ob, Wobf, bo, out);
}
